// MoE_53515292508963
// MI455X (gfx1250) — compile-verified, matrix operand order checked by reading, not run on hardware
//
#include <hip/hip_runtime.h>

// ---------------- problem constants (from reference) ----------------
constexpr int T = 4096;   // tokens
constexpr int D = 2048;   // hidden
constexpr int F = 1024;   // ffn dim
constexpr int E = 8;      // experts

// ---------------- tiling ----------------
constexpr int M_TILE  = 32;             // tokens per block (2 WMMA m-tiles)
constexpr int BW      = 64;             // N-slab width (4 waves x 16 cols)
constexpr int K_CHUNK = 64;             // K rows staged per TDM tile
constexpr int SX_S    = D + 8;          // LDS stride (halfs) for X tile
constexpr int SH_S    = F + 8;          // LDS stride (halfs) for H tile
constexpr int SX_BYTES   = M_TILE * SX_S * 2;       // 131584
constexpr int SH_BYTES   = M_TILE * SH_S * 2;       //  66048
constexpr int BUF_FLOATS = K_CHUNK * BW;            //  4096 floats = 16 KB
constexpr int BUFW_BYTES = 4 * BUF_FLOATS * 4;      //  65536 (4 staging regions)
constexpr int SMEM_BYTES = SX_BYTES + SH_BYTES + BUFW_BYTES + 256; // ~257 KB

typedef __attribute__((ext_vector_type(8)))  float        v8f;
typedef __attribute__((ext_vector_type(16))) __bf16       v16bf;
typedef __attribute__((ext_vector_type(8)))  __bf16       v8bf;
typedef __attribute__((ext_vector_type(4)))  unsigned int u32x4;
typedef __attribute__((ext_vector_type(8)))  int          i32x8;
typedef __attribute__((ext_vector_type(4)))  int          i32x4;

// -------------------------------------------------------------------
// Tensor Data Mover: DMA a tileH x tileW fp32 tile (row-major, leading
// dim `width`) from global memory into LDS.  D# per CDNA5 ISA:
//   group0: count=1 | lds_addr | global_addr[56:0] | type=2
//   group1: data_size=4B | tensor_dim0/1 | tile_dim0/1 | dim0 stride
// Issued by one wave; tracked with TENSORcnt.
// -------------------------------------------------------------------
__device__ __forceinline__ void tdm_load_tile_f32(unsigned ldsAddr, const float* gptr,
                                                  unsigned width, unsigned rowsRemain,
                                                  unsigned tileW, unsigned tileH) {
    const unsigned long long ga = (unsigned long long)(uintptr_t)gptr;
    u32x4 g0;
    g0[0] = 1u;                                            // count=1 (user D#)
    g0[1] = ldsAddr;                                       // lds_addr (bytes)
    g0[2] = (unsigned)(ga & 0xffffffffu);                  // global_addr[31:0]
    g0[3] = (unsigned)((ga >> 32) & 0x1ffffffu) | (2u << 30); // addr[56:32] | type=2
    i32x8 g1;
    g1[0] = (int)(2u << 16);                               // data_size = 4 bytes
    g1[1] = (int)(width << 16);                            // tensor_dim0[15:0]
    g1[2] = (int)((width >> 16) | (rowsRemain << 16));     // dim0[31:16] | dim1[15:0]
    g1[3] = (int)(((rowsRemain >> 16) & 0xffffu) | (tileW << 16)); // dim1[31:16]|tile_dim0
    g1[4] = (int)(tileH & 0xffffu);                        // tile_dim1 (tile_dim2=0)
    g1[5] = (int)width;                                    // tensor_dim0_stride[31:0]
    g1[6] = 0;                                             // stride0[47:32] | stride1 lo
    g1[7] = 0;
    i32x4 z4 = {0, 0, 0, 0};
#if __clang_major__ >= 23
    i32x8 z8 = {0, 0, 0, 0, 0, 0, 0, 0};
    __builtin_amdgcn_tensor_load_to_lds(g0, g1, z4, z4, z8, 0);
#else
    __builtin_amdgcn_tensor_load_to_lds(g0, g1, z4, z4, 0);
#endif
}

__device__ __forceinline__ unsigned lds_addr_of(const void* p) {
    return (unsigned)(uintptr_t)p;   // flat LDS address: low 32 bits = LDS offset
}

// -------------------------------------------------------------------
// A-fragment (16x32 bf16) from LDS, per CDNA5 WMMA layout: two 16B ds loads.
// -------------------------------------------------------------------
__device__ __forceinline__ v16bf load_a_frag(const __bf16* rowBase, int stride,
                                             int lane, int kBase) {
    const int l15 = lane & 15;
    const int kh  = (lane & 16) ? 8 : 0;
    const __bf16* p = rowBase + l15 * stride + kBase + kh;
    v8bf lo = *(const v8bf*)p;          // k = kh .. kh+7
    v8bf hi = *(const v8bf*)(p + 16);   // k = kh+16 .. kh+23
    return __builtin_shufflevector(lo, hi, 0,1,2,3,4,5,6,7,8,9,10,11,12,13,14,15);
}

// -------------------------------------------------------------------
// B-fragment (32x16) from a TDM-staged LDS fp32 tile (32+ rows x BW cols),
// converted to bf16 at read time.
// -------------------------------------------------------------------
__device__ __forceinline__ v16bf load_b_frag_lds(const float* base, int lane, int nSub) {
    const int col = nSub * 16 + (lane & 15);
    const int kh  = (lane & 16) ? 8 : 0;
    const float* p = base + kh * BW + col;
    v16bf b;
#pragma unroll
    for (int j = 0; j < 8; ++j) {
        b[j]     = (__bf16)p[j * BW];
        b[j + 8] = (__bf16)p[(j + 16) * BW];
    }
    return b;
}

// ===================================================================
// Kernel 1: router.  One wave per token -> 8 logits, softmax, top-2,
// renormalize, scatter (token, weight) into per-expert lists.
// ===================================================================
__global__ __launch_bounds__(256)
void router_topk_kernel(const float* __restrict__ x, const float* __restrict__ rk,
                        int* __restrict__ cnt, int* __restrict__ tokList,
                        float* __restrict__ wList) {
    const int lane = threadIdx.x & 31;
    const int wid  = threadIdx.x >> 5;
    const int t    = blockIdx.x * 8 + wid;
    if (t >= T) return;

    float acc[E];
#pragma unroll
    for (int e = 0; e < E; ++e) acc[e] = 0.f;

    const float* xr = x + (size_t)t * D;
    for (int d = lane; d < D; d += 32) {
        const float xv = xr[d];
#pragma unroll
        for (int e = 0; e < E; ++e) acc[e] += xv * rk[d * E + e];
    }
#pragma unroll
    for (int e = 0; e < E; ++e)
        for (int off = 16; off > 0; off >>= 1)
            acc[e] += __shfl_xor(acc[e], off, 32);

    if (lane == 0) {
        float mx = acc[0];
#pragma unroll
        for (int e = 1; e < E; ++e) mx = fmaxf(mx, acc[e]);
        float p[E], s = 0.f;
#pragma unroll
        for (int e = 0; e < E; ++e) { p[e] = __expf(acc[e] - mx); s += p[e]; }
        const float inv = 1.f / s;
#pragma unroll
        for (int e = 0; e < E; ++e) p[e] *= inv;

        int e0 = 0; float p0 = p[0];
#pragma unroll
        for (int e = 1; e < E; ++e) if (p[e] > p0) { p0 = p[e]; e0 = e; }
        int e1 = (e0 == 0) ? 1 : 0; float p1 = p[e1];
#pragma unroll
        for (int e = 0; e < E; ++e)
            if (e != e0 && p[e] > p1) { p1 = p[e]; e1 = e; }

        const float w0 = 1.f / (1.f + __expf(p1 - p0));   // softmax over top-2
        const float w1 = 1.f - w0;

        int s0 = atomicAdd(&cnt[e0], 1);
        tokList[e0 * T + s0] = t;  wList[e0 * T + s0] = w0;
        int s1 = atomicAdd(&cnt[e1], 1);
        tokList[e1 * T + s1] = t;  wList[e1 * T + s1] = w1;
    }
}

// ===================================================================
// Kernel 2: per-expert grouped MLP with TDM-staged weights.
// Block = 256 threads (8 waves = 2 M-tiles x 4 N-tiles), 32 tokens.
//   phase 0: gather 32 token rows of x -> LDS bf16
//   phase 1: G=X*Wg, U=X*Wu (WMMA bf16; Wg/Wu tiles DMA'd by TDM,
//            double-buffered, s_wait_tensorcnt pipelined); H=silu(G)*U -> LDS
//   phase 2: O=H*Wd (TDM-staged Wd); out[token] += weight * O (f32 atomics)
// ===================================================================
__global__ __launch_bounds__(256)
void moe_expert_kernel(const float* __restrict__ x,
                       const float* __restrict__ wg,
                       const float* __restrict__ wu,
                       const float* __restrict__ wd,
                       const int*   __restrict__ cnt,
                       const int*   __restrict__ tokList,
                       const float* __restrict__ wList,
                       float* __restrict__ out) {
    extern __shared__ __align__(16) char smem[];
    __bf16* sX     = (__bf16*)smem;                                   // [32][SX_S]
    __bf16* sH     = (__bf16*)(smem + SX_BYTES);                      // [32][SH_S]
    float*  bufW   = (float*)(smem + SX_BYTES + SH_BYTES);            // 4 x 16KB staging
    int*    tokLds = (int*)  (smem + SX_BYTES + SH_BYTES + BUFW_BYTES);
    float*  wLds   = (float*)(smem + SX_BYTES + SH_BYTES + BUFW_BYTES + 128);

    const int tid  = threadIdx.x;
    const int lane = tid & 31;
    const int wid  = tid >> 5;
    const int e    = blockIdx.y;

    const int cntE  = cnt[e];
    const int tile0 = blockIdx.x * M_TILE;
    if (tile0 >= cntE) return;   // block-uniform early exit

    if (tid < M_TILE) {
        const int slot = tile0 + tid;
        if (slot < cntE) { tokLds[tid] = tokList[e * T + slot]; wLds[tid] = wList[e * T + slot]; }
        else             { tokLds[tid] = 0;                     wLds[tid] = 0.f; }
    }
    __syncthreads();

    // ---- phase 0: gather X tile -> LDS (bf16) ----
    for (int idx = tid; idx < M_TILE * D; idx += 256) {
        const int r = idx >> 11;            // / D
        const int c = idx & (D - 1);
        sX[r * SX_S + c] = (__bf16)x[(size_t)tokLds[r] * D + c];
    }
    __syncthreads();

    const int mBase = (wid & 1) * 16;
    const int nSub  = wid >> 1;
    const int half8 = (lane & 16) ? 8 : 0;
    const int l15   = lane & 15;

    const float* Wg = wg + (size_t)e * D * F;
    const float* Wu = wu + (size_t)e * D * F;

    // ---- phase 1: gate & up projections, fused SiLU ----
    constexpr int NC1 = D / K_CHUNK;   // 32 K-chunks
    for (int nIter = 0; nIter < F / BW; ++nIter) {
        const int nSlab = nIter * BW;
        if (wid == 0) {   // prologue: DMA chunk 0 (G into region 0, U into region 1)
            tdm_load_tile_f32(lds_addr_of(bufW),              Wg + nSlab, F, D, BW, K_CHUNK);
            tdm_load_tile_f32(lds_addr_of(bufW + BUF_FLOATS), Wu + nSlab, F, D, BW, K_CHUNK);
        }
        v8f accG = {}; v8f accU = {};
        for (int ic = 0; ic < NC1; ++ic) {
            const int kb = ic * K_CHUNK;
            const float* bG = bufW + (ic & 1) * 2 * BUF_FLOATS;
            const float* bU = bG + BUF_FLOATS;
            if (wid == 0) {
                if (ic + 1 < NC1) {   // DMA next chunk into the other buffer pair
                    float* nG = bufW + ((ic + 1) & 1) * 2 * BUF_FLOATS;
                    const int kn = kb + K_CHUNK;
                    tdm_load_tile_f32(lds_addr_of(nG),              Wg + (size_t)kn * F + nSlab,
                                      F, D - kn, BW, K_CHUNK);
                    tdm_load_tile_f32(lds_addr_of(nG + BUF_FLOATS), Wu + (size_t)kn * F + nSlab,
                                      F, D - kn, BW, K_CHUNK);
                    __builtin_amdgcn_s_wait_tensorcnt(2);   // retire chunk ic's pair
                } else {
                    __builtin_amdgcn_s_wait_tensorcnt(0);
                }
            }
            __syncthreads();          // chunk ic visible to all waves
#pragma unroll
            for (int sub = 0; sub < 2; ++sub) {
                v16bf a  = load_a_frag(sX + mBase * SX_S, SX_S, lane, kb + sub * 32);
                v16bf fG = load_b_frag_lds(bG + sub * 32 * BW, lane, nSub);
                v16bf fU = load_b_frag_lds(bU + sub * 32 * BW, lane, nSub);
                accG = __builtin_amdgcn_wmma_f32_16x16x32_bf16(false, a, false, fG,
                                                               (short)0, accG, false, false);
                accU = __builtin_amdgcn_wmma_f32_16x16x32_bf16(false, a, false, fU,
                                                               (short)0, accU, false, false);
            }
            __syncthreads();          // all readers done before buffers are overwritten
        }
#pragma unroll
        for (int i = 0; i < 8; ++i) {
            const float g = accG[i];
            const float u = accU[i];
            const float h = (g / (1.f + __expf(-g))) * u;   // silu(g)*u
            sH[(mBase + half8 + i) * SH_S + nSlab + nSub * 16 + l15] = (__bf16)h;
        }
    }
    __syncthreads();   // sH complete; staging buffers free

    // ---- phase 2: down projection + weighted scatter-add ----
    const float* Wd = wd + (size_t)e * F * D;
    constexpr int NC2 = F / K_CHUNK;   // 16 K-chunks
    for (int n2 = 0; n2 < D / BW; ++n2) {
        const int nSlab = n2 * BW;
        if (wid == 0)
            tdm_load_tile_f32(lds_addr_of(bufW), Wd + nSlab, D, F, BW, K_CHUNK);
        v8f acc = {};
        for (int ic = 0; ic < NC2; ++ic) {
            const int kb = ic * K_CHUNK;
            const float* bD = bufW + (ic & 1) * BUF_FLOATS;
            if (wid == 0) {
                if (ic + 1 < NC2) {
                    float* nD = bufW + ((ic + 1) & 1) * BUF_FLOATS;
                    const int kn = kb + K_CHUNK;
                    tdm_load_tile_f32(lds_addr_of(nD), Wd + (size_t)kn * D + nSlab,
                                      D, F - kn, BW, K_CHUNK);
                    __builtin_amdgcn_s_wait_tensorcnt(1);
                } else {
                    __builtin_amdgcn_s_wait_tensorcnt(0);
                }
            }
            __syncthreads();
#pragma unroll
            for (int sub = 0; sub < 2; ++sub) {
                v16bf a = load_a_frag(sH + mBase * SH_S, SH_S, lane, kb + sub * 32);
                v16bf b = load_b_frag_lds(bD + sub * 32 * BW, lane, nSub);
                acc = __builtin_amdgcn_wmma_f32_16x16x32_bf16(false, a, false, b,
                                                              (short)0, acc, false, false);
            }
            __syncthreads();
        }
#pragma unroll
        for (int i = 0; i < 8; ++i) {
            const int   row = mBase + half8 + i;
            const float v   = acc[i] * wLds[row];           // padded rows: weight 0
            unsafeAtomicAdd(&out[(size_t)tokLds[row] * D + nSlab + nSub * 16 + l15], v);
        }
    }
}

// ===================================================================
extern "C" void kernel_launch(void* const* d_in, const int* in_sizes, int n_in,
                              void* d_out, int out_size, void* d_ws, size_t ws_size,
                              hipStream_t stream) {
    const float* x  = (const float*)d_in[0];   // (T, D)
    const float* rk = (const float*)d_in[1];   // (D, E)
    const float* wg = (const float*)d_in[2];   // (E, D, F)
    const float* wu = (const float*)d_in[3];   // (E, D, F)
    const float* wd = (const float*)d_in[4];   // (E, F, D)
    float* out = (float*)d_out;                // (T, D)

    char* ws = (char*)d_ws;
    int*   cnt     = (int*)ws;                               // E counters
    int*   tokList = (int*)(ws + 256);                       // E*T ints
    float* wList   = (float*)(ws + 256 + (size_t)E * T * 4); // E*T floats

    hipMemsetAsync(cnt, 0, E * sizeof(int), stream);
    hipMemsetAsync(out, 0, (size_t)T * D * sizeof(float), stream);

    router_topk_kernel<<<T / 8, 256, 0, stream>>>(x, rk, cnt, tokList, wList);

    moe_expert_kernel<<<dim3(T / M_TILE, E), 256, SMEM_BYTES, stream>>>(
        x, wg, wu, wd, cnt, tokList, wList, out);
}
